// MultiHeadAttention_62345745269382
// MI455X (gfx1250) — matrix operand order repaired and checked by reading, not compiled, not run on hardware
//
#include <hip/hip_runtime.h>
#include <stdint.h>

#define D_MODEL  1024
#define N_HEADS  16
#define HEAD_DIM 64
#define SEQ_LEN  2048
#define BATCH    2
#define ROWS     (BATCH * SEQ_LEN)   // 4096

typedef __attribute__((ext_vector_type(16))) __bf16        v16bf;
typedef __attribute__((ext_vector_type(8)))  float         v8f;
typedef __attribute__((ext_vector_type(4)))  unsigned int  u32x4;
typedef __attribute__((ext_vector_type(4)))  unsigned int  ui32x4;
typedef __attribute__((ext_vector_type(8)))  int           i32x8;
typedef __attribute__((ext_vector_type(4)))  int           i32x4;

#if __has_builtin(__builtin_amdgcn_tensor_load_to_lds)
#define HAVE_TDM 1
#else
#define HAVE_TDM 0
#endif

// ---------- bf16 helpers (RNE) ----------
__device__ __forceinline__ unsigned short f2bf(float f) {
  union { float f; unsigned int u; } x; x.f = f;
  unsigned int lsb = (x.u >> 16) & 1u;
  x.u += 0x7FFFu + lsb;
  return (unsigned short)(x.u >> 16);
}
__device__ __forceinline__ float bf2f(unsigned short h) {
  union { unsigned int u; float f; } x; x.u = ((unsigned int)h) << 16;
  return x.f;
}
__device__ __forceinline__ v8f vzero8() {
  v8f z;
#pragma unroll
  for (int i = 0; i < 8; ++i) z[i] = 0.f;
  return z;
}

// A-operand frag (16x32 MxK, 16-bit): lane half sel=lane>>4 holds K chunks
// {base..base+7} and {base+16..base+23} where base = k0 + sel*8.
__device__ __forceinline__ v16bf load_a_frag(const unsigned short* p) {
  union { u32x4 u[2]; v16bf v; } r;
  r.u[0] = *reinterpret_cast<const u32x4*>(p);
  r.u[1] = *reinterpret_cast<const u32x4*>(p + 16);
  return r.v;
}
// B-operand frag (32x16 KxN, 16-bit): lane = column n; per-lane K chunks
// {base..base+7} and {base+16..base+23} where base = k0 + sel*8.
__device__ __forceinline__ v16bf load_b_frag(const unsigned short* p) {
  union { u32x4 u[2]; v16bf v; } r;
  r.u[0] = *reinterpret_cast<const u32x4*>(p);
  r.u[1] = *reinterpret_cast<const u32x4*>(p + 16);
  return r.v;
}

// ---------- gfx1250 async global->LDS (ASYNCcnt) ----------
// Low 32 bits of a generic pointer to __shared__ == LDS byte offset.
__device__ __forceinline__ unsigned lds_off(const void* p) {
  return (unsigned)(uintptr_t)p;
}
__device__ __forceinline__ void async_load_b128(unsigned lds_addr, unsigned voff,
                                                const void* sbase) {
  asm volatile("global_load_async_to_lds_b128 %0, %1, %2"
               :: "v"(lds_addr), "v"(voff), "s"(sbase)
               : "memory");
}
__device__ __forceinline__ void wait_asynccnt0() {
  asm volatile("s_wait_asynccnt 0x0" ::: "memory");
}

#if HAVE_TDM
// ---------- gfx1250 Tensor Data Mover: 2D tile (rows x kelems, 2B elems) ----------
// D# per CDNA5 ISA: group0 = {flags, lds_addr, global_addr, type=2},
// group1 = {data_size, tensor dims, tile dims, dim0 stride}. Remaining groups zero
// (2D tile: tile_dim2/3/4 = 0 -> unused). This toolchain's builtin takes 6 args.
__device__ __forceinline__ void tdm_load_tile_2d(const void* gaddr, unsigned lds_addr,
                                                 int rows, int kelems,
                                                 int stride_elems) {
  unsigned long long ga = (unsigned long long)gaddr;
  ui32x4 g0;
  g0[0] = 1u;                                              // count=1, no gather
  g0[1] = lds_addr;                                        // LDS byte address
  g0[2] = (unsigned)(ga & 0xFFFFFFFFu);                    // global_addr[31:0]
  g0[3] = (unsigned)((ga >> 32) & 0x1FFFFFFu) | 0x80000000u; // [56:32] | type=2
  i32x8 g1;
  g1[0] = 0x00010000;                       // data_size=1 -> 2 bytes
  g1[1] = (int)((unsigned)kelems << 16);    // tensor_dim0[15:0] (== tile_dim0)
  g1[2] = (int)((unsigned)rows << 16);      // tensor_dim1[15:0] (== tile_dim1)
  g1[3] = (int)((unsigned)kelems << 16);    // tile_dim0
  g1[4] = rows;                             // tile_dim1 (tile_dim2 = 0)
  g1[5] = stride_elems;                     // tensor_dim0_stride[31:0]
  g1[6] = 0;
  g1[7] = 0;
  i32x4 gz4 = {0, 0, 0, 0};
  i32x8 gz8 = {0, 0, 0, 0, 0, 0, 0, 0};
  __builtin_amdgcn_tensor_load_to_lds(g0, g1, gz4, gz4, gz8, 0);
}
#endif

// ---------- fp32 -> bf16 convert ----------
__global__ void cvt_f32_bf16(const float* __restrict__ in,
                             unsigned short* __restrict__ out, int n) {
  int i = blockIdx.x * blockDim.x + threadIdx.x;
  if (i < n) out[i] = f2bf(in[i]);
}

// ---------- GEMM: C[M,N] = A[M,K](bf16) * W[N,K](bf16)^T + bias ----------
// wave: 16x64 tile; block: 8 waves stacked in M -> 128x64; grid (M/128, N/64).
// W stage = 64 rows x 64 k (8KB), double-buffered, filled by the TDM
// (tensor_load_to_lds, TENSORcnt) or async-to-LDS fallback.
__global__ __launch_bounds__(256) void gemm_xwT(
    const unsigned short* __restrict__ A,
    const unsigned short* __restrict__ W,
    const float* __restrict__ bias,
    float* __restrict__ outF,            // fp32 output (or null)
    unsigned short* __restrict__ outB,   // bf16 output (or null)
    int M, int N, int K) {
  (void)M;
  __shared__ __align__(16) unsigned short wtile[2][64][64];

  const int tid  = threadIdx.x;
  const int lane = tid & 31;
  const int wave = tid >> 5;
  const int sel  = lane >> 4;
  const int ln   = lane & 15;
  const int m0   = blockIdx.x * 128 + wave * 16;
  const int nb   = blockIdx.y * 64;

#if !HAVE_TDM
  // async fallback: 512 b128 chunks per stage; 2 chunks per thread
  const int frow0 = tid >> 2;                 // 0..63
  const int fkc0  = (tid & 3) * 8;            // 0..24
  const int fkc1  = fkc0 + 32;                // 32..56
#endif

  const unsigned short* arow = A + (size_t)(m0 + ln) * K;

  v8f acc[4];
#pragma unroll
  for (int t = 0; t < 4; ++t) acc[t] = vzero8();

  // ---- prologue: fill stage 0 (k=0), preload A frags for k=0,32 ----
#if HAVE_TDM
  if (wave == 0)
    tdm_load_tile_2d(W + (size_t)nb * K, lds_off(&wtile[0][0][0]), 64, 64, K);
#else
  async_load_b128(lds_off(&wtile[0][frow0][fkc0]),
                  (unsigned)(((nb + frow0) * K + fkc0) * 2), W);
  async_load_b128(lds_off(&wtile[0][frow0][fkc1]),
                  (unsigned)(((nb + frow0) * K + fkc1) * 2), W);
#endif
  v16bf va0 = load_a_frag(arow + 0 + sel * 8);
  v16bf va1 = load_a_frag(arow + 32 + sel * 8);

  for (int k = 0; k < K; k += 64) {
    const int st = (k >> 6) & 1;
#if HAVE_TDM
    __builtin_amdgcn_s_wait_tensorcnt(0);
#else
    wait_asynccnt0();
#endif
    __syncthreads();   // stage visible to all waves; old stage reads retired
    if (k + 64 < K) {
#if HAVE_TDM
      if (wave == 0)
        tdm_load_tile_2d(W + (size_t)nb * K + (k + 64),
                         lds_off(&wtile[st ^ 1][0][0]), 64, 64, K);
#else
      async_load_b128(lds_off(&wtile[st ^ 1][frow0][fkc0]),
                      (unsigned)(((nb + frow0) * K + (k + 64) + fkc0) * 2), W);
      async_load_b128(lds_off(&wtile[st ^ 1][frow0][fkc1]),
                      (unsigned)(((nb + frow0) * K + (k + 64) + fkc1) * 2), W);
#endif
    }

    v16bf va0n = va0, va1n = va1;
    if (k + 64 < K) {
      va0n = load_a_frag(arow + (k + 64) + sel * 8);
      va1n = load_a_frag(arow + (k + 96) + sel * 8);
    }

#pragma unroll
    for (int t = 0; t < 4; ++t) {
      v16bf vb = load_b_frag(&wtile[st][t * 16 + ln][sel * 8]);
      acc[t] = __builtin_amdgcn_wmma_f32_16x16x32_bf16(
          false, va0, false, vb, (short)0, acc[t], false, false);
    }
#pragma unroll
    for (int t = 0; t < 4; ++t) {
      v16bf vb = load_b_frag(&wtile[st][t * 16 + ln][32 + sel * 8]);
      acc[t] = __builtin_amdgcn_wmma_f32_16x16x32_bf16(
          false, va1, false, vb, (short)0, acc[t], false, false);
    }
    va0 = va0n; va1 = va1n;
  }

#pragma unroll
  for (int t = 0; t < 4; ++t) {
    int col = nb + t * 16 + ln;
    float bv = bias ? bias[col] : 0.f;
#pragma unroll
    for (int r = 0; r < 8; ++r) {
      int row = m0 + r + sel * 8;            // C layout: VGPR r -> M = r + 8*sel
      float v = acc[t][r] + bv;
      if (outF) outF[(size_t)row * N + col] = v;
      else      outB[(size_t)row * N + col] = f2bf(v);
    }
  }
}

// ---------- RoPE in place on Q and K ([B*L][D] bf16) ----------
__global__ void rope_inplace(unsigned short* Qb, unsigned short* Kb) {
  const int per = BATCH * SEQ_LEN * N_HEADS * 32;   // 2^21
  int idx = blockIdx.x * blockDim.x + threadIdx.x;
  if (idx >= 2 * per) return;
  unsigned short* p = (idx < per) ? Qb : Kb;
  int i = (idx < per) ? idx : idx - per;
  int d  = i & 31;
  int hh = (i >> 5) & 15;
  int l  = (i >> 9) & 2047;
  int bb = i >> 20;
  size_t base = ((size_t)(bb * SEQ_LEN + l) * D_MODEL) + hh * HEAD_DIM + d;
  float v0 = bf2f(p[base]);
  float v1 = bf2f(p[base + 32]);
  float ang = (float)l * __powf(10000.f, -(float)d * (1.f / 32.f));
  float s, c;
  __sincosf(ang, &s, &c);
  p[base]      = f2bf(v0 * c - v1 * s);
  p[base + 32] = f2bf(v1 * c + v0 * s);
}

// ---------- V transpose: [B*L][D] -> [B][H][hd][L] ----------
__global__ void transpose_v(const unsigned short* __restrict__ Vb,
                            unsigned short* __restrict__ Vt) {
  const int total = BATCH * N_HEADS * HEAD_DIM * SEQ_LEN;  // 2^22
  int idx = blockIdx.x * blockDim.x + threadIdx.x;
  if (idx >= total) return;
  int l  = idx & 2047;
  int d  = (idx >> 11) & 63;
  int hh = (idx >> 17) & 15;
  int bb = idx >> 21;
  Vt[idx] = Vb[((size_t)(bb * SEQ_LEN + l) * D_MODEL) + hh * HEAD_DIM + d];
}

// ---------- flash-style attention ----------
// grid.x = B * H * (L/128); block = 256 = 8 waves; wave handles 16 q-rows.
// K tile (32x64) and V^T tile (64x32) staged via async-to-LDS, double buffered;
// all 8 waves share them (same head).
__global__ __launch_bounds__(256) void attn_fwd(
    const unsigned short* __restrict__ Q,    // [B*L][D] bf16 (rope'd)
    const unsigned short* __restrict__ Km,   // [B*L][D] bf16 (rope'd)
    const unsigned short* __restrict__ Vt,   // [B][H][hd][L] bf16
    unsigned short* __restrict__ O) {        // [B*L][D] bf16
  __shared__ __align__(16) unsigned short kt[2][32][64];   // [key][hd]
  __shared__ __align__(16) unsigned short vts[2][64][32];  // [d][l]
  __shared__ __align__(16) unsigned short pls[8][16][40];  // per-wave P slab

  const int tid  = threadIdx.x;
  const int lane = tid & 31;
  const int wave = tid >> 5;
  const int sel  = lane >> 4;
  const int ln   = lane & 15;

  const int qblk  = blockIdx.x & 15;         // L/128 = 16
  const int h     = (blockIdx.x >> 4) & 15;
  const int b     = blockIdx.x >> 8;
  const int qbase = qblk * 128 + wave * 16;

  // Q tile, kept in registers: two A-frags covering hd = 64
  const unsigned short* qrow =
      Q + ((size_t)(b * SEQ_LEN + qbase + ln) * D_MODEL) + h * HEAD_DIM;
  v16bf qa0 = load_a_frag(qrow + 0 + sel * 8);
  v16bf qa1 = load_a_frag(qrow + 32 + sel * 8);

  // async fill assignments (256 threads fill 2 x 4KB per step)
  const unsigned short* Kghead =
      Km + (size_t)b * SEQ_LEN * D_MODEL + h * HEAD_DIM;
  const int krow = tid >> 3;            // 0..31 key row
  const int kc8  = (tid & 7) * 8;       // hd offset 0..56
  const unsigned kvoff = (unsigned)((krow * D_MODEL + kc8) * 2);
  const unsigned klds0 = lds_off(&kt[0][krow][kc8]);
  const unsigned klds1 = lds_off(&kt[1][krow][kc8]);

  const unsigned short* Vghead =
      Vt + (size_t)(b * N_HEADS + h) * HEAD_DIM * SEQ_LEN;
  const int vrow = tid >> 2;            // 0..63 d
  const int vc8  = (tid & 3) * 8;       // l offset 0..24
  const unsigned vvoff = (unsigned)((vrow * SEQ_LEN + vc8) * 2);
  const unsigned vlds0 = lds_off(&vts[0][vrow][vc8]);
  const unsigned vlds1 = lds_off(&vts[1][vrow][vc8]);

  float mrow[8], lrow[8];
#pragma unroll
  for (int r = 0; r < 8; ++r) { mrow[r] = -3.0e38f; lrow[r] = 0.f; }
  v8f ot[4];
#pragma unroll
  for (int t = 0; t < 4; ++t) ot[t] = vzero8();

  // prefetch first K/V tiles
  async_load_b128(klds0, kvoff, Kghead);
  async_load_b128(vlds0, vvoff, Vghead);

  for (int kk = 0; kk < SEQ_LEN; kk += 32) {
    const int cur = (kk >> 5) & 1;
    wait_asynccnt0();
    __syncthreads();
    if (kk + 32 < SEQ_LEN) {
      async_load_b128(cur ? klds0 : klds1,
                      kvoff + (unsigned)((kk + 32) * D_MODEL * 2), Kghead);
      async_load_b128(cur ? vlds0 : vlds1,
                      vvoff + (unsigned)((kk + 32) * 2), Vghead);
    }

    // ---- scores S = Q * K^T for 32 keys (B-frags from LDS) ----
    v16bf b00 = load_b_frag(&kt[cur][ln][sel * 8]);
    v16bf b01 = load_b_frag(&kt[cur][ln][32 + sel * 8]);
    v16bf b10 = load_b_frag(&kt[cur][16 + ln][sel * 8]);
    v16bf b11 = load_b_frag(&kt[cur][16 + ln][32 + sel * 8]);
    v8f s0 = vzero8(), s1 = vzero8();
    s0 = __builtin_amdgcn_wmma_f32_16x16x32_bf16(false, qa0, false, b00, (short)0, s0, false, false);
    s0 = __builtin_amdgcn_wmma_f32_16x16x32_bf16(false, qa1, false, b01, (short)0, s0, false, false);
    s1 = __builtin_amdgcn_wmma_f32_16x16x32_bf16(false, qa0, false, b10, (short)0, s1, false, false);
    s1 = __builtin_amdgcn_wmma_f32_16x16x32_bf16(false, qa1, false, b11, (short)0, s1, false, false);

    // ---- online softmax (fp32); row lives in 16 lanes of one half ----
    float p0[8], p1[8], rescale[8];
#pragma unroll
    for (int r = 0; r < 8; ++r) {
      float a = s0[r] * 0.125f;   // 1/sqrt(64)
      float c = s1[r] * 0.125f;
      float mx = fmaxf(a, c);
#pragma unroll
      for (int off = 1; off < 16; off <<= 1)
        mx = fmaxf(mx, __shfl_xor(mx, off, 32));
      float mn = fmaxf(mrow[r], mx);
      rescale[r] = __expf(mrow[r] - mn);
      p0[r] = __expf(a - mn);
      p1[r] = __expf(c - mn);
      float rs = p0[r] + p1[r];
#pragma unroll
      for (int off = 1; off < 16; off <<= 1)
        rs += __shfl_xor(rs, off, 32);
      lrow[r] = lrow[r] * rescale[r] + rs;
      mrow[r] = mn;
    }
#pragma unroll
    for (int t = 0; t < 4; ++t)
#pragma unroll
      for (int r = 0; r < 8; ++r) ot[t][r] = ot[t][r] * rescale[r];

    // ---- P (C-layout fp32) -> LDS bf16 -> A-layout frag ----
    // Per-wave slab; same-wave DS ops execute in order -> no barrier needed.
#pragma unroll
    for (int r = 0; r < 8; ++r) {
      int row = r + sel * 8;
      pls[wave][row][ln]      = f2bf(p0[r]);
      pls[wave][row][16 + ln] = f2bf(p1[r]);
    }
    v16bf pa;
    {
      const unsigned short* lp = &pls[wave][ln][sel * 8];
      union { u32x4 u[2]; v16bf v; } r;
      r.u[0] = *reinterpret_cast<const u32x4*>(lp);
      r.u[1] = *reinterpret_cast<const u32x4*>(lp + 16);
      pa = r.v;
    }

    // ---- O += P(16x32) * V(32x64): 4 WMMAs, B-frags from LDS ----
#pragma unroll
    for (int t = 0; t < 4; ++t) {
      v16bf vb = load_b_frag(&vts[cur][t * 16 + ln][sel * 8]);
      ot[t] = __builtin_amdgcn_wmma_f32_16x16x32_bf16(
          false, pa, false, vb, (short)0, ot[t], false, false);
    }
  }

  // ---- epilogue: normalize and store bf16 ----
#pragma unroll
  for (int t = 0; t < 4; ++t) {
#pragma unroll
    for (int r = 0; r < 8; ++r) {
      int row = qbase + r + sel * 8;
      int col = h * HEAD_DIM + t * 16 + ln;
      float v = ot[t][r] / lrow[r];
      O[(size_t)(b * SEQ_LEN + row) * D_MODEL + col] = f2bf(v);
    }
  }
}

// ---------------------------------------------------------------------------
extern "C" void kernel_launch(void* const* d_in, const int* in_sizes, int n_in,
                              void* d_out, int out_size, void* d_ws, size_t ws_size,
                              hipStream_t stream) {
  (void)in_sizes; (void)n_in; (void)out_size; (void)ws_size;
  const float* x  = (const float*)d_in[0];
  const float* qw = (const float*)d_in[1];
  const float* qb = (const float*)d_in[2];
  const float* kw = (const float*)d_in[3];
  const float* kb = (const float*)d_in[4];
  const float* vw = (const float*)d_in[5];
  const float* vbi= (const float*)d_in[6];
  const float* ow = (const float*)d_in[7];
  const float* ob = (const float*)d_in[8];
  float* out = (float*)d_out;

  uint8_t* ws = (uint8_t*)d_ws;
  const size_t MB = 1024u * 1024u;
  unsigned short* xb  = (unsigned short*)(ws + 0);        // 8 MB (reused as Obuf)
  unsigned short* wqb = (unsigned short*)(ws + 8 * MB);   // 2 MB
  unsigned short* wkb = (unsigned short*)(ws + 10 * MB);
  unsigned short* wvb = (unsigned short*)(ws + 12 * MB);
  unsigned short* wob = (unsigned short*)(ws + 14 * MB);
  unsigned short* Qb  = (unsigned short*)(ws + 16 * MB);  // 8 MB
  unsigned short* Kb  = (unsigned short*)(ws + 24 * MB);  // 8 MB
  unsigned short* Vb  = (unsigned short*)(ws + 32 * MB);  // 8 MB
  unsigned short* Vt  = (unsigned short*)(ws + 40 * MB);  // 8 MB -> 48 MB total
  unsigned short* Ob  = xb;  // alias: x no longer needed after QKV GEMMs

  const int nx = ROWS * D_MODEL;       // 4M
  const int nw = D_MODEL * D_MODEL;    // 1M
  cvt_f32_bf16<<<(nx + 255) / 256, 256, 0, stream>>>(x,  xb,  nx);
  cvt_f32_bf16<<<(nw + 255) / 256, 256, 0, stream>>>(qw, wqb, nw);
  cvt_f32_bf16<<<(nw + 255) / 256, 256, 0, stream>>>(kw, wkb, nw);
  cvt_f32_bf16<<<(nw + 255) / 256, 256, 0, stream>>>(vw, wvb, nw);
  cvt_f32_bf16<<<(nw + 255) / 256, 256, 0, stream>>>(ow, wob, nw);

  dim3 ggrid(ROWS / 128, D_MODEL / 64);  // (32, 16)
  gemm_xwT<<<ggrid, 256, 0, stream>>>(xb, wqb, qb, nullptr, Qb, ROWS, D_MODEL, D_MODEL);
  gemm_xwT<<<ggrid, 256, 0, stream>>>(xb, wkb, kb, nullptr, Kb, ROWS, D_MODEL, D_MODEL);
  gemm_xwT<<<ggrid, 256, 0, stream>>>(xb, wvb, vbi, nullptr, Vb, ROWS, D_MODEL, D_MODEL);

  const int nr = 2 * BATCH * SEQ_LEN * N_HEADS * 32;  // Q and K halves
  rope_inplace<<<(nr + 255) / 256, 256, 0, stream>>>(Qb, Kb);

  const int nt = BATCH * N_HEADS * HEAD_DIM * SEQ_LEN;
  transpose_v<<<(nt + 255) / 256, 256, 0, stream>>>(Vb, Vt);

  attn_fwd<<<BATCH * N_HEADS * (SEQ_LEN / 128), 256, 0, stream>>>(Qb, Kb, Vt, Ob);

  gemm_xwT<<<ggrid, 256, 0, stream>>>(Ob, wob, ob, out, nullptr, ROWS, D_MODEL, D_MODEL);
}
